// TransformerModel1_51960514347243
// MI455X (gfx1250) — matrix operand order repaired and checked by reading, not compiled, not run on hardware
//
#include <hip/hip_runtime.h>
#include <math.h>

#define B 32
#define S 512
#define H 768
#define NHEAD 8
#define DH 96
#define E 4096
#define NC 2
#define NEGV (-1000000000.0f)
#define EPSV 1e-5f
#define NEDGE (E + S)            // 4608 edges per batch (incl. self loops)
#define TOT_EDGE (B * NEDGE)     // 147456

typedef float v2f  __attribute__((ext_vector_type(2)));
typedef float v4f  __attribute__((ext_vector_type(4)));
typedef float v8f  __attribute__((ext_vector_type(8)));
typedef __bf16 v8bf  __attribute__((ext_vector_type(8)));
typedef __bf16 v16bf __attribute__((ext_vector_type(16)));
typedef unsigned short us4 __attribute__((ext_vector_type(4)));

// RNE float -> bf16 split: x ~= hi + lo (bit-twiddled, no __bf16 arithmetic)
__device__ inline void bfsplit(float x, unsigned short& h, unsigned short& l) {
    unsigned u = __float_as_uint(x);
    unsigned short hb = (unsigned short)((u + 0x7FFFu + ((u >> 16) & 1u)) >> 16);
    float hf = __uint_as_float((unsigned)hb << 16);
    unsigned v = __float_as_uint(x - hf);
    unsigned short lb = (unsigned short)((v + 0x7FFFu + ((v >> 16) & 1u)) >> 16);
    h = hb; l = lb;
}

// two contiguous 16B LDS reads -> one 16-element bf16 fragment
__device__ inline v16bf ldfrag(const unsigned short* p0, const unsigned short* p1) {
    v8bf a = *(const v8bf*)p0;
    v8bf b = *(const v8bf*)p1;
    return __builtin_shufflevector(a, b, 0,1,2,3,4,5,6,7,8,9,10,11,12,13,14,15);
}

// ---------------------------------------------------------------------------
// bf16x3-split WMMA GEMM: C[M,N] = A[M,K] @ W[K,N] + bias[N] (+resid)(+ReLU)
// 256 threads = 8 waves; tile 128(M) x 64(N); K chunked by 32.
// Per chunk/tile: 3x v_wmma_f32_16x16x32_bf16 (AhiBhi + AloBhi + AhiBlo),
// fp32 accumulation -> ~fp32 accuracy at the 16-bit matrix-pipe rate.
// ---------------------------------------------------------------------------
template<bool RELU>
__global__ __launch_bounds__(256)
void gemm_wmma_bf16x3(const float* __restrict__ A, const float* __restrict__ W,
                      const float* __restrict__ bias, const float* __restrict__ resid,
                      float* __restrict__ C, int M, int K, int N)
{
    // row stride 40 ushorts = 80B (multiple of 16B for ds b128 loads)
    __shared__ unsigned short AsH[128][40];
    __shared__ unsigned short AsL[128][40];
    __shared__ unsigned short WsH[64][40];
    __shared__ unsigned short WsL[64][40];

    const int tid  = threadIdx.x;
    const int wave = tid >> 5;
    const int lane = tid & 31;
    const int l16  = lane & 15;
    const int hi   = lane >> 4;                  // 0 or 1

    const int nTile = blockIdx.x * 64;
    const int mTile = blockIdx.y * 128;

    v8f acc[4];
    #pragma unroll
    for (int t = 0; t < 4; ++t)
        #pragma unroll
        for (int r = 0; r < 8; ++r) acc[t][r] = 0.0f;

    for (int k0 = 0; k0 < K; k0 += 32) {
        // --- stage A tile 128x32: each thread converts 4x float4 -> hi/lo ---
        {
            int r  = tid >> 3;              // 0..31
            int kc = (tid & 7) << 2;        // 0..28
            #pragma unroll
            for (int i = 0; i < 4; ++i) {
                int row  = r + 32 * i;
                int grow = mTile + row;
                grow = grow < M ? grow : (M - 1);   // clamp; store is guarded
                v4f v = *(const v4f*)(A + (size_t)grow * K + k0 + kc);
                us4 hh, ll;
                #pragma unroll
                for (int j = 0; j < 4; ++j) {
                    unsigned short h, l;
                    bfsplit(v[j], h, l);
                    hh[j] = h; ll[j] = l;
                }
                *(us4*)&AsH[row][kc] = hh;
                *(us4*)&AsL[row][kc] = ll;
            }
        }
        // --- stage W tile transposed: Ws[n][k], 32k x 64n ---
        {
            int n  = tid & 63;
            int kb = tid >> 6;              // 0..3
            #pragma unroll
            for (int i = 0; i < 8; ++i) {
                int kk = kb + 4 * i;
                unsigned short h, l;
                bfsplit(W[(size_t)(k0 + kk) * N + nTile + n], h, l);
                WsH[n][kk] = h;
                WsL[n][kk] = l;
            }
        }
        __syncthreads();
        // --- fragments (16-bit A 16x32 / B 32x16 wave32 layouts) ---
        {
            int m = wave * 16 + l16;
            v16bf ahi = ldfrag(&AsH[m][8 * hi], &AsH[m][16 + 8 * hi]);
            v16bf alo = ldfrag(&AsL[m][8 * hi], &AsL[m][16 + 8 * hi]);
            #pragma unroll
            for (int t = 0; t < 4; ++t) {
                int n = t * 16 + l16;
                v16bf bhi = ldfrag(&WsH[n][8 * hi], &WsH[n][16 + 8 * hi]);
                v16bf blo = ldfrag(&WsL[n][8 * hi], &WsL[n][16 + 8 * hi]);
                acc[t] = __builtin_amdgcn_wmma_f32_16x16x32_bf16(
                    false, ahi, false, bhi, (short)0, acc[t], false, false);
                acc[t] = __builtin_amdgcn_wmma_f32_16x16x32_bf16(
                    false, alo, false, bhi, (short)0, acc[t], false, false);
                acc[t] = __builtin_amdgcn_wmma_f32_16x16x32_bf16(
                    false, ahi, false, blo, (short)0, acc[t], false, false);
            }
        }
        __syncthreads();
    }
    // --- epilogue: C/D layout -> VGPR r: lanes0-15 M=r, lanes16-31 M=r+8 ---
    const int mAdd = hi << 3;
    #pragma unroll
    for (int t = 0; t < 4; ++t) {
        int gn = nTile + t * 16 + l16;
        float bv = bias[gn];
        #pragma unroll
        for (int r = 0; r < 8; ++r) {
            int gm = mTile + wave * 16 + r + mAdd;
            if (gm < M) {
                float v = acc[t][r] + bv;
                if (resid) v += resid[(size_t)gm * N + gn];
                if (RELU)  v = v > 0.0f ? v : 0.0f;
                C[(size_t)gm * N + gn] = v;
            }
        }
    }
}

// ---------------------------------------------------------------------------
// GAT pieces
// ---------------------------------------------------------------------------
__device__ inline void atomicMaxF(float* addr, float v) {
    if (v >= 0.0f) atomicMax((int*)addr, __float_as_int(v));
    else           atomicMin((unsigned int*)addr, __float_as_uint(v));
}

__global__ void init_md_kernel(float* emax, float* denom) {
    int i = blockIdx.x * blockDim.x + threadIdx.x;
    if (i < B * S) { emax[i] = NEGV; denom[i] = 0.0f; }
}

__global__ void init_gout_kernel(const float* __restrict__ enc,
                                 const float* __restrict__ bias,
                                 float* __restrict__ out, int n) {
    int i = blockIdx.x * blockDim.x + threadIdx.x;
    if (i < n) out[i] = enc[i] + bias[i % H];   // gat_bias + residual encoded
}

// one wave per edge: e = leakyrelu(xl[src]+xr[dst], 0.2) . att ; seg-max
__global__ __launch_bounds__(256)
void gat_edge_e(const float* __restrict__ xl, const float* __restrict__ xr,
                const float* __restrict__ att, const int* __restrict__ eidx,
                float* __restrict__ ebuf, int* __restrict__ sdbuf,
                float* __restrict__ emax)
{
    int gid  = blockIdx.x * 8 + (threadIdx.x >> 5);
    int lane = threadIdx.x & 31;
    if (gid >= TOT_EDGE) return;
    int b = gid / NEDGE, j = gid % NEDGE;
    int s, d, valid;
    if (j < E) {
        s = eidx[(size_t)b * 2 * E + j];
        d = eidx[(size_t)b * 2 * E + E + j];
        valid = ((unsigned)s < S) && ((unsigned)d < S);
        if (!valid) { s = 0; d = 0; }
    } else { s = j - E; d = s; valid = 1; }
    const float* pl = xl + ((size_t)b * S + s) * H;
    const float* pr = xr + ((size_t)b * S + d) * H;
    float sum = 0.0f;
    for (int i = lane; i < H; i += 32) {
        float t = pl[i] + pr[i];
        t = t > 0.0f ? t : 0.2f * t;
        sum += t * att[i];
    }
    #pragma unroll
    for (int off = 16; off > 0; off >>= 1) sum += __shfl_xor(sum, off, 32);
    if (lane == 0) {
        float ev = valid ? sum : NEGV;
        ebuf[gid]  = ev;
        sdbuf[gid] = s | (d << 10) | (valid << 20);
        atomicMaxF(&emax[b * S + d], ev);
    }
}

__global__ void gat_edge_p(const float* __restrict__ ebuf, const int* __restrict__ sdbuf,
                           const float* __restrict__ emax, float* __restrict__ pbuf,
                           float* __restrict__ denom)
{
    int gid = blockIdx.x * blockDim.x + threadIdx.x;
    if (gid >= TOT_EDGE) return;
    int b = gid / NEDGE;
    int sd = sdbuf[gid];
    int d = (sd >> 10) & 1023;
    int valid = (sd >> 20) & 1;
    float p = valid ? __expf(ebuf[gid] - emax[b * S + d]) : 0.0f;
    pbuf[gid] = p;
    atomicAdd(&denom[b * S + d], p);
}

__global__ __launch_bounds__(256)
void gat_scatter(const float* __restrict__ xl, const int* __restrict__ sdbuf,
                 const float* __restrict__ pbuf, const float* __restrict__ denom,
                 float* __restrict__ out)
{
    int gid = blockIdx.x;    // edge id
    int b = gid / NEDGE;
    int sd = sdbuf[gid];
    int s = sd & 1023, d = (sd >> 10) & 1023;
    float alpha = pbuf[gid] / denom[b * S + d];
    const float* src = xl + ((size_t)b * S + s) * H;
    float* dst = out + ((size_t)b * S + d) * H;
    for (int f = threadIdx.x; f < H; f += 256)
        atomicAdd(&dst[f], alpha * src[f]);
}

// ---------------------------------------------------------------------------
// Attention (only query position 0 matters downstream)
// ---------------------------------------------------------------------------
__global__ void gather_row0(const float* __restrict__ g, float* __restrict__ x0) {
    int i = blockIdx.x * blockDim.x + threadIdx.x;
    if (i < B * H) {
        int b = i / H, f = i % H;
        x0[i] = g[(size_t)b * S * H + f];
    }
}

__global__ __launch_bounds__(256)
void attn_q0(const float* __restrict__ q, const float* __restrict__ kbuf,
             const float* __restrict__ vbuf, const int* __restrict__ mask,
             float* __restrict__ ctx)
{
    __shared__ float qs[DH];
    __shared__ float sc[S];
    __shared__ float red[256];
    int bh = blockIdx.x;
    int b = bh / NHEAD, h = bh % NHEAD;
    int tid = threadIdx.x;
    if (tid < DH) qs[tid] = q[(size_t)b * H + h * DH + tid];
    __syncthreads();
    const float scale = 0.102062072615966f;   // 1/sqrt(96)
    float lmax = -3.4e38f;
    for (int j = tid; j < S; j += 256) {
        const float* kp = kbuf + ((size_t)b * S + j) * H + h * DH;
        float sdot = 0.0f;
        for (int d = 0; d < DH; ++d) sdot += qs[d] * kp[d];
        sdot *= scale;
        if (mask[b * S + j] == 0) sdot = NEGV;
        sc[j] = sdot;
        lmax = fmaxf(lmax, sdot);
    }
    red[tid] = lmax; __syncthreads();
    for (int st = 128; st > 0; st >>= 1) {
        if (tid < st) red[tid] = fmaxf(red[tid], red[tid + st]);
        __syncthreads();
    }
    float m = red[0]; __syncthreads();
    float lsum = 0.0f;
    for (int j = tid; j < S; j += 256) {
        float pz = __expf(sc[j] - m);
        sc[j] = pz;
        lsum += pz;
    }
    red[tid] = lsum; __syncthreads();
    for (int st = 128; st > 0; st >>= 1) {
        if (tid < st) red[tid] += red[tid + st];
        __syncthreads();
    }
    float inv = 1.0f / red[0]; __syncthreads();
    for (int d = tid; d < DH; d += 256) {
        float a = 0.0f;
        const float* vp = vbuf + (size_t)b * S * H + h * DH + d;
        for (int j = 0; j < S; ++j) a += sc[j] * vp[(size_t)j * H];
        ctx[(size_t)b * H + h * DH + d] = a * inv;
    }
}

// ---------------------------------------------------------------------------
// LayerNorm over H per row (32 rows), and final fc2
// ---------------------------------------------------------------------------
__global__ __launch_bounds__(256)
void layernorm_row(const float* __restrict__ x, const float* __restrict__ g,
                   const float* __restrict__ bta, float* __restrict__ out)
{
    __shared__ float red[256];
    int b = blockIdx.x, tid = threadIdx.x;
    const float* row = x + (size_t)b * H;
    float s1 = 0.0f;
    for (int i = tid; i < H; i += 256) s1 += row[i];
    red[tid] = s1; __syncthreads();
    for (int st = 128; st > 0; st >>= 1) { if (tid < st) red[tid] += red[tid + st]; __syncthreads(); }
    float mu = red[0] / (float)H; __syncthreads();
    float s2 = 0.0f;
    for (int i = tid; i < H; i += 256) { float d = row[i] - mu; s2 += d * d; }
    red[tid] = s2; __syncthreads();
    for (int st = 128; st > 0; st >>= 1) { if (tid < st) red[tid] += red[tid + st]; __syncthreads(); }
    float rstd = rsqrtf(red[0] / (float)H + EPSV);
    for (int i = tid; i < H; i += 256)
        out[(size_t)b * H + i] = (row[i] - mu) * rstd * g[i] + bta[i];
}

__global__ void fc2_kernel(const float* __restrict__ x, const float* __restrict__ w,
                           const float* __restrict__ bias, float* __restrict__ out)
{
    int i = blockIdx.x * blockDim.x + threadIdx.x;
    if (i >= B * NC) return;
    int b = i / NC, c = i % NC;
    float s = bias[c];
    for (int k = 0; k < 128; ++k) s += x[b * 128 + k] * w[k * NC + c];
    out[i] = s;
}

// ---------------------------------------------------------------------------
extern "C" void kernel_launch(void* const* d_in, const int* in_sizes, int n_in,
                              void* d_out, int out_size, void* d_ws, size_t ws_size,
                              hipStream_t stream)
{
    const float* diff_enc = (const float*)d_in[0];
    const float* msg_enc  = (const float*)d_in[1];
    const int*   msg_mask = (const int*)  d_in[2];
    const int*   eidx_diff= (const int*)  d_in[3];
    const int*   eidx_msg = (const int*)  d_in[4];
    const float* gat_wl   = (const float*)d_in[5];
    const float* gat_bl   = (const float*)d_in[6];
    const float* gat_wr   = (const float*)d_in[7];
    const float* gat_br   = (const float*)d_in[8];
    const float* gat_att  = (const float*)d_in[9];
    const float* gat_bias = (const float*)d_in[10];
    const float* wq = (const float*)d_in[11]; const float* bq = (const float*)d_in[12];
    const float* wk = (const float*)d_in[13]; const float* bk = (const float*)d_in[14];
    const float* wv = (const float*)d_in[15]; const float* bv = (const float*)d_in[16];
    const float* wo = (const float*)d_in[17]; const float* bo = (const float*)d_in[18];
    const float* ln_g = (const float*)d_in[19]; const float* ln_b = (const float*)d_in[20];
    const float* fc0_w = (const float*)d_in[21]; const float* fc0_b = (const float*)d_in[22];
    const float* fc1_w = (const float*)d_in[23]; const float* fc1_b = (const float*)d_in[24];
    const float* fc2_w = (const float*)d_in[25]; const float* fc2_b = (const float*)d_in[26];

    const size_t BIG = (size_t)B * S * H;       // 12,582,912 floats
    float* ws    = (float*)d_ws;
    float* bufA  = ws;                 // xl, later k
    float* bufB  = ws + BIG;           // xr, later v
    float* diffg = ws + 2 * BIG;
    float* msgg  = ws + 3 * BIG;
    float* sm    = ws + 4 * BIG;
    float* ebuf  = sm;                  sm += TOT_EDGE;
    float* pbuf  = sm;                  sm += TOT_EDGE;
    int*   sdbuf = (int*)sm;            sm += TOT_EDGE;
    float* emax  = sm;                  sm += B * S;
    float* denom = sm;                  sm += B * S;
    float* x0    = sm;                  sm += B * H;
    float* qb    = sm;                  sm += B * H;
    float* ctxb  = sm;                  sm += B * H;
    float* aout  = sm;                  sm += B * H;
    float* fused = sm;                  sm += B * H;
    float* h0    = sm;                  sm += B * 512;
    float* h1    = sm;                  sm += B * 128;

    const int M = B * S;                         // 16384
    dim3 blk(256);
    dim3 gBig(H / 64, (M + 127) / 128);          // 12 x 128

    // ---- GAT for both streams (shared weights) ----
    for (int st = 0; st < 2; ++st) {
        const float* enc  = st == 0 ? diff_enc  : msg_enc;
        const int*   eidx = st == 0 ? eidx_diff : eidx_msg;
        float*       gout = st == 0 ? diffg     : msgg;

        gemm_wmma_bf16x3<false><<<gBig, blk, 0, stream>>>(enc, gat_wl, gat_bl, nullptr, bufA, M, H, H);
        gemm_wmma_bf16x3<false><<<gBig, blk, 0, stream>>>(enc, gat_wr, gat_br, nullptr, bufB, M, H, H);
        init_md_kernel<<<(B * S + 255) / 256, blk, 0, stream>>>(emax, denom);
        init_gout_kernel<<<(int)((BIG + 255) / 256), blk, 0, stream>>>(enc, gat_bias, gout, (int)BIG);
        gat_edge_e<<<(TOT_EDGE + 7) / 8, blk, 0, stream>>>(bufA, bufB, gat_att, eidx, ebuf, sdbuf, emax);
        gat_edge_p<<<(TOT_EDGE + 255) / 256, blk, 0, stream>>>(ebuf, sdbuf, emax, pbuf, denom);
        gat_scatter<<<TOT_EDGE, blk, 0, stream>>>(bufA, sdbuf, pbuf, denom, gout);
    }

    // ---- k, v projections over full msg_g (reuse big buffers) ----
    gemm_wmma_bf16x3<false><<<gBig, blk, 0, stream>>>(msgg, wk, bk, nullptr, bufA, M, H, H);  // k
    gemm_wmma_bf16x3<false><<<gBig, blk, 0, stream>>>(msgg, wv, bv, nullptr, bufB, M, H, H);  // v

    // ---- only query position 0 is consumed downstream ----
    gather_row0<<<(B * H + 255) / 256, blk, 0, stream>>>(diffg, x0);
    dim3 gRow(H / 64, 1);
    gemm_wmma_bf16x3<false><<<gRow, blk, 0, stream>>>(x0, wq, bq, nullptr, qb, B, H, H);
    attn_q0<<<B * NHEAD, blk, 0, stream>>>(qb, bufA, bufB, msg_mask, ctxb);
    gemm_wmma_bf16x3<false><<<gRow, blk, 0, stream>>>(ctxb, wo, bo, x0, aout, B, H, H);
    layernorm_row<<<B, blk, 0, stream>>>(aout, ln_g, ln_b, fused);

    // ---- MLP head ----
    dim3 gFc0(512 / 64, 1), gFc1(128 / 64, 1);
    gemm_wmma_bf16x3<true><<<gFc0, blk, 0, stream>>>(fused, fc0_w, fc0_b, nullptr, h0, B, H, 512);
    gemm_wmma_bf16x3<true><<<gFc1, blk, 0, stream>>>(h0, fc1_w, fc1_b, nullptr, h1, B, 512, 128);
    fc2_kernel<<<1, 64, 0, stream>>>(h1, fc2_w, fc2_b, (float*)d_out);
}
